// RelativeMultiHeadSelfAttentionBlock_2482491097355
// MI455X (gfx1250) — matrix operand order repaired and checked by reading, not compiled, not run on hardware
//
#include <hip/hip_runtime.h>

typedef unsigned short u16;
typedef __attribute__((ext_vector_type(16))) __bf16 v16bf;
typedef __attribute__((ext_vector_type(8)))  __bf16 bf16x8;
typedef __attribute__((ext_vector_type(8)))  float  v8f;
typedef __attribute__((ext_vector_type(8)))  u16    u16x8;
typedef __attribute__((ext_vector_type(4)))  float  f32x4;

union ABf { v16bf v; u16x8 h[2]; };

// native bf16 convert: lowers to v_cvt (packed) instead of bit-twiddling VALU
__device__ __forceinline__ u16 f2bf(float f) {
    __bf16 h = (__bf16)f;
    return __builtin_bit_cast(u16, h);
}

__device__ __forceinline__ v8f wmma_bf16(const ABf& a, const ABf& b, v8f c) {
    return __builtin_amdgcn_wmma_f32_16x16x32_bf16(
        /*neg_a=*/false, a.v, /*neg_b=*/false, b.v,
        /*c_mod=*/(short)0, c, /*reuse_a=*/false, /*reuse_b=*/false);
}

// ---------------------------------------------------------------------------
// Bulk fp32 -> bf16 conversion: 8 elements/thread, 32B in / 16B out.
// Runs once per tensor so the GEMM inner loops are pure bf16 loads + WMMA.
// ---------------------------------------------------------------------------
__global__ __launch_bounds__(256)
void cvt_bf16_kernel(const float* __restrict__ src, u16* __restrict__ dst, int n8) {
    int i = blockIdx.x * 256 + threadIdx.x;
    if (i < n8) {
        f32x4 a = *(const f32x4*)(src + (size_t)i * 8);
        f32x4 b = *(const f32x4*)(src + (size_t)i * 8 + 4);
        bf16x8 r;
        r[0] = (__bf16)a[0]; r[1] = (__bf16)a[1]; r[2] = (__bf16)a[2]; r[3] = (__bf16)a[3];
        r[4] = (__bf16)b[0]; r[5] = (__bf16)b[1]; r[6] = (__bf16)b[2]; r[7] = (__bf16)b[3];
        *(u16x8*)(dst + (size_t)i * 8) = __builtin_bit_cast(u16x8, r);
    }
}

// ---------------------------------------------------------------------------
// C[M,512] = A[M,512] @ W[512,512]^T + bias + bias2   (bf16 in, bf16 out)
// mode 0: row-major bf16 [M,512]     mode 1: V-transposed bf16 [B,H,64,1024]
// One wave computes a 16x64 tile; K loop in steps of 32; pure b128 loads.
// ---------------------------------------------------------------------------
__global__ __launch_bounds__(32)
void proj_gemm_kernel(const u16* __restrict__ A, const u16* __restrict__ W,
                      const float* __restrict__ bias, const float* __restrict__ bias2,
                      u16* __restrict__ out, int M, int mode) {
    const int lane = threadIdx.x;
    const int lm = lane & 15, lh = lane >> 4;
    const int m0 = blockIdx.x * 16;
    const int n0 = blockIdx.y * 64;
    int row = m0 + lm; if (row > M - 1) row = M - 1;     // clamp for M=8188 tail
    const u16* arow = A + (size_t)row * 512;

    v8f acc[4] = {};
    for (int kc = 0; kc < 512; kc += 32) {
        ABf a;
        a.h[0] = *(const u16x8*)(arow + kc + lh * 16);       // K = lh*16 + 0..7
        a.h[1] = *(const u16x8*)(arow + kc + lh * 16 + 8);   // K = lh*16 + 8..15
        #pragma unroll
        for (int j = 0; j < 4; ++j) {
            const u16* wrow = W + (size_t)(n0 + j * 16 + lm) * 512 + kc + lh * 16;
            ABf b;                                           // K = lh*16 + 0..15
            b.h[0] = *(const u16x8*)wrow;
            b.h[1] = *(const u16x8*)(wrow + 8);
            acc[j] = wmma_bf16(a, b, acc[j]);
        }
    }
    #pragma unroll
    for (int j = 0; j < 4; ++j) {
        const int n = n0 + j * 16 + lm;
        float badd = (bias ? bias[n] : 0.f) + (bias2 ? bias2[n] : 0.f);
        #pragma unroll
        for (int v = 0; v < 8; ++v) {
            int r = m0 + v + (lh << 3);
            if (r < M) {
                float val = acc[j][v] + badd;
                if (mode == 0) {
                    out[(size_t)r * 512 + n] = f2bf(val);
                } else {                                     // Vt[b][h][dk][t]
                    int b_ = r >> 10, t = r & 1023;
                    int hh = n >> 6, d = n & 63;
                    out[(size_t)((b_ * 8 + hh) * 64 + d) * 1024 + t] = f2bf(val);
                }
            }
        }
    }
}

// ---------------------------------------------------------------------------
// Fused rel-pos attention, flash-style. One wave per (b, h, 16-row t-tile).
// scores = (Qu·K^T + rel_shift(Qv·R^T)) / 8 ; mask -> 1e-4 ; online softmax ;
// O += P·V ; head-interleaved bf16 output for the final projection.
// ---------------------------------------------------------------------------
__global__ __launch_bounds__(32)
void rel_attn_kernel(const u16* __restrict__ Qu, const u16* __restrict__ Qv,
                     const u16* __restrict__ Kb, const u16* __restrict__ Vt,
                     const u16* __restrict__ Rb, const unsigned char* __restrict__ mask,
                     u16* __restrict__ Cc) {
    const int lane = threadIdx.x;
    const int lm = lane & 15, lh = lane >> 4;
    const int t0 = blockIdx.x * 16;
    const int h  = blockIdx.y;
    const int b  = blockIdx.z;

    __shared__ __align__(16) u16 pbuf[16 * 48];          // P relayout scratch (wave-private)

    // Preload Q operands (A layout), resident all kernel: 16x64 each, 2 k-chunks
    ABf qu[2], qv[2];
    {
        const u16* qub = Qu + (size_t)(b * 1024 + t0 + lm) * 512 + h * 64;
        const u16* qvb = Qv + (size_t)(b * 1024 + t0 + lm) * 512 + h * 64;
        #pragma unroll
        for (int kc = 0; kc < 2; ++kc) {
            qu[kc].h[0] = *(const u16x8*)(qub + kc * 32 + lh * 16);
            qu[kc].h[1] = *(const u16x8*)(qub + kc * 32 + lh * 16 + 8);
            qv[kc].h[0] = *(const u16x8*)(qvb + kc * 32 + lh * 16);
            qv[kc].h[1] = *(const u16x8*)(qvb + kc * 32 + lh * 16 + 8);
        }
    }

    v8f O[4] = {};
    float rowmax[8], lsum[8];
    #pragma unroll
    for (int v = 0; v < 8; ++v) { rowmax[v] = -3.4e38f; lsum[v] = 0.f; }

    const u16* kbase = Kb + (size_t)b * 1024 * 512 + h * 64;
    const u16* rbase = Rb + (size_t)b * 2047 * 512 + h * 64;
    const u16* vbase = Vt + (size_t)((b * 8 + h) * 64) * 1024;
    const unsigned char* mrow = mask + b * 1024;
    const int halfSel = lane & 16;

    for (int s0 = 0; s0 < 1024; s0 += 32) {
        // ---- content scores: two 16x16 tiles over 32 key columns ----
        v8f sc[2] = {};
        #pragma unroll
        for (int kc = 0; kc < 2; ++kc) {
            #pragma unroll
            for (int j = 0; j < 2; ++j) {
                const u16* kr = kbase + (size_t)(s0 + j * 16 + lm) * 512 + kc * 32 + lh * 16;
                ABf bm; bm.h[0] = *(const u16x8*)kr; bm.h[1] = *(const u16x8*)(kr + 8);
                sc[j] = wmma_bf16(qu[kc], bm, sc[j]);
            }
        }
        // ---- rpe band: l in [s0-t0+1008, +1054] -> three 16-wide l-tiles ----
        const int lbase = s0 - t0 + 1008;
        v8f rp[3] = {};
        #pragma unroll
        for (int jt = 0; jt < 3; ++jt) {
            int l = lbase + jt * 16 + lm;
            if (l > 2046) l = 2046;                       // clamped row never consumed
            const u16* rr0 = rbase + (size_t)l * 512;
            #pragma unroll
            for (int kc = 0; kc < 2; ++kc) {
                const u16* rr = rr0 + kc * 32 + lh * 16;
                ABf bm; bm.h[0] = *(const u16x8*)rr; bm.h[1] = *(const u16x8*)(rr + 8);
                rp[jt] = wmma_bf16(qv[kc], bm, rp[jt]);
            }
        }
        if (s0 + 32 < 1024)                               // hint next K rows into cache
            __builtin_prefetch(kbase + (size_t)(s0 + 32 + lm) * 512, 0, 0);

        // ---- rel-shift gather (lane permute), scale, mask ----
        int msk0 = mrow[s0 + lm];
        int msk1 = mrow[s0 + 16 + lm];
        #pragma unroll
        for (int j = 0; j < 2; ++j) {
            #pragma unroll
            for (int v = 0; v < 8; ++v) {
                int m = v + (lh << 3);
                int c = lm + j * 16 - m + 15;             // 0..46 within the band
                int srcLane = (c & 15) | halfSel;         // same half, same VGPR index
                float r0 = __shfl(rp[0][v], srcLane, 32);
                float r1 = __shfl(rp[1][v], srcLane, 32);
                float r2 = __shfl(rp[2][v], srcLane, 32);
                float rel = (c < 16) ? r0 : ((c < 32) ? r1 : r2);
                float val = (sc[j][v] + rel) * 0.125f;    // 1/sqrt(64)
                if ((j ? msk1 : msk0) != 0) val = 1e-4f;
                sc[j][v] = val;
            }
        }

        // ---- online softmax (rows live in 16-lane halves) ----
        #pragma unroll
        for (int v = 0; v < 8; ++v) {
            float mv = fmaxf(sc[0][v], sc[1][v]);
            #pragma unroll
            for (int d = 8; d >= 1; d >>= 1) mv = fmaxf(mv, __shfl_xor(mv, d, 32));
            float nm = fmaxf(rowmax[v], mv);
            float corr = __expf(rowmax[v] - nm);
            rowmax[v] = nm;
            float p0 = __expf(sc[0][v] - nm);
            float p1 = __expf(sc[1][v] - nm);
            sc[0][v] = p0; sc[1][v] = p1;
            float rs = p0 + p1;
            #pragma unroll
            for (int d = 8; d >= 1; d >>= 1) rs += __shfl_xor(rs, d, 32);
            lsum[v] = lsum[v] * corr + rs;
            #pragma unroll
            for (int j2 = 0; j2 < 4; ++j2) O[j2][v] *= corr;
        }

        // ---- P: C-layout -> A-layout via LDS ----
        #pragma unroll
        for (int j = 0; j < 2; ++j)
            #pragma unroll
            for (int v = 0; v < 8; ++v) {
                int m = v + (lh << 3);
                pbuf[m * 48 + j * 16 + lm] = f2bf(sc[j][v]);
            }
        asm volatile("s_wait_dscnt 0x0" ::: "memory");
        ABf pa;
        pa.h[0] = *(const u16x8*)&pbuf[lm * 48 + lh * 16];
        pa.h[1] = *(const u16x8*)&pbuf[lm * 48 + lh * 16 + 8];

        // ---- O += P @ V  (V pre-transposed: Vt rows are s-contiguous) ----
        #pragma unroll
        for (int j2 = 0; j2 < 4; ++j2) {
            const u16* vr = vbase + (size_t)(j2 * 16 + lm) * 1024 + s0 + lh * 16;
            ABf bm; bm.h[0] = *(const u16x8*)vr; bm.h[1] = *(const u16x8*)(vr + 8);
            O[j2] = wmma_bf16(pa, bm, O[j2]);
        }
    }

    // ---- normalize + head-interleaved store (reference's transpose-free view) ----
    #pragma unroll
    for (int v = 0; v < 8; ++v) {
        float inv = 1.0f / lsum[v];
        int t = t0 + v + (lh << 3);
        int tau = h * 128 + (t >> 3);
        int colbase = (t & 7) * 64;
        #pragma unroll
        for (int j2 = 0; j2 < 4; ++j2) {
            int dk = j2 * 16 + lm;
            Cc[(size_t)(b * 1024 + tau) * 512 + colbase + dk] = f2bf(O[j2][v] * inv);
        }
    }
}

// ---------------------------------------------------------------------------
// out[4096,512] (fp32) = concat_bf16 @ Wo_bf16^T + bo
// ---------------------------------------------------------------------------
__global__ __launch_bounds__(32)
void out_gemm_kernel(const u16* __restrict__ A, const u16* __restrict__ W,
                     const float* __restrict__ bias, float* __restrict__ out) {
    const int lane = threadIdx.x;
    const int lm = lane & 15, lh = lane >> 4;
    const int m0 = blockIdx.x * 16;
    const int n0 = blockIdx.y * 64;
    const u16* arow = A + (size_t)(m0 + lm) * 512;
    v8f acc[4] = {};
    for (int kc = 0; kc < 512; kc += 32) {
        ABf a;
        a.h[0] = *(const u16x8*)(arow + kc + lh * 16);
        a.h[1] = *(const u16x8*)(arow + kc + lh * 16 + 8);
        #pragma unroll
        for (int j = 0; j < 4; ++j) {
            const u16* wrow = W + (size_t)(n0 + j * 16 + lm) * 512 + kc + lh * 16;
            ABf b;
            b.h[0] = *(const u16x8*)wrow;
            b.h[1] = *(const u16x8*)(wrow + 8);
            acc[j] = wmma_bf16(a, b, acc[j]);
        }
    }
    #pragma unroll
    for (int j = 0; j < 4; ++j) {
        int n = n0 + j * 16 + lm;
        float badd = bias[n];
        #pragma unroll
        for (int v = 0; v < 8; ++v) {
            int r = m0 + v + (lh << 3);
            out[(size_t)r * 512 + n] = acc[j][v] + badd;
        }
    }
}

extern "C" void kernel_launch(void* const* d_in, const int* in_sizes, int n_in,
                              void* d_out, int out_size, void* d_ws, size_t ws_size,
                              hipStream_t stream) {
    const float* query = (const float*)d_in[0];
    const float* key   = (const float*)d_in[1];
    const float* value = (const float*)d_in[2];
    const float* pos   = (const float*)d_in[3];
    const unsigned char* mask = (const unsigned char*)d_in[4];  // jax bool = 1 byte
    const float* Wq = (const float*)d_in[5];
    const float* bq = (const float*)d_in[6];
    const float* Wk = (const float*)d_in[7];
    const float* bk = (const float*)d_in[8];
    const float* Wv = (const float*)d_in[9];
    const float* bv = (const float*)d_in[10];
    const float* Wr = (const float*)d_in[11];
    const float* ub = (const float*)d_in[12];  // u_bias flat [H*DK] == per-column
    const float* vb = (const float*)d_in[13];  // v_bias flat [H*DK]
    const float* Wo = (const float*)d_in[14];
    const float* bo = (const float*)d_in[15];
    float* out = (float*)d_out;

    // bf16 workspace (~53 MB total)
    const size_t NA = 4096 * 512;       // activation tensor elems
    const size_t NP = 8188 * 512;       // pos-embedding elems
    const size_t NW = 512 * 512;        // weight elems
    u16* Qu  = (u16*)d_ws;              // projected q + u_bias
    u16* Qv  = Qu  + NA;                // projected q + v_bias
    u16* Kp  = Qv  + NA;                // projected k
    u16* Vt  = Kp  + NA;                // projected v, transposed [B,H,64,T]
    u16* Rp  = Vt  + NA;                // projected r
    u16* Cc  = Rp  + NP;                // attention output, head-interleaved
    u16* Xq  = Cc  + NA;                // bf16 copies of inputs
    u16* Xk  = Xq  + NA;
    u16* Xv  = Xk  + NA;
    u16* Xp  = Xv  + NA;
    u16* Wqb = Xp  + NP;                // bf16 copies of weights
    u16* Wkb = Wqb + NW;
    u16* Wvb = Wkb + NW;
    u16* Wrb = Wvb + NW;
    u16* Wob = Wrb + NW;

    dim3 blk32(32);
    // ---- stage 0: one-time fp32 -> bf16 conversions ----
    int a8 = (int)(NA / 8), p8 = (int)(NP / 8), w8 = (int)(NW / 8);
    cvt_bf16_kernel<<<dim3((a8 + 255) / 256), 256, 0, stream>>>(query, Xq, a8);
    cvt_bf16_kernel<<<dim3((a8 + 255) / 256), 256, 0, stream>>>(key,   Xk, a8);
    cvt_bf16_kernel<<<dim3((a8 + 255) / 256), 256, 0, stream>>>(value, Xv, a8);
    cvt_bf16_kernel<<<dim3((p8 + 255) / 256), 256, 0, stream>>>(pos,   Xp, p8);
    cvt_bf16_kernel<<<dim3((w8 + 255) / 256), 256, 0, stream>>>(Wq, Wqb, w8);
    cvt_bf16_kernel<<<dim3((w8 + 255) / 256), 256, 0, stream>>>(Wk, Wkb, w8);
    cvt_bf16_kernel<<<dim3((w8 + 255) / 256), 256, 0, stream>>>(Wv, Wvb, w8);
    cvt_bf16_kernel<<<dim3((w8 + 255) / 256), 256, 0, stream>>>(Wr, Wrb, w8);
    cvt_bf16_kernel<<<dim3((w8 + 255) / 256), 256, 0, stream>>>(Wo, Wob, w8);

    // ---- stage 1: projections (all WMMA) ----
    proj_gemm_kernel<<<dim3(256, 8), blk32, 0, stream>>>(Xq, Wqb, bq, ub,      Qu, 4096, 0);
    proj_gemm_kernel<<<dim3(256, 8), blk32, 0, stream>>>(Xq, Wqb, bq, vb,      Qv, 4096, 0);
    proj_gemm_kernel<<<dim3(256, 8), blk32, 0, stream>>>(Xk, Wkb, bk, nullptr, Kp, 4096, 0);
    proj_gemm_kernel<<<dim3(256, 8), blk32, 0, stream>>>(Xv, Wvb, bv, nullptr, Vt, 4096, 1);
    proj_gemm_kernel<<<dim3(512, 8), blk32, 0, stream>>>(Xp, Wrb, nullptr, nullptr, Rp, 8188, 0);

    // ---- stage 2: fused rel-pos flash attention ----
    rel_attn_kernel<<<dim3(64, 8, 4), blk32, 0, stream>>>(Qu, Qv, Kp, Vt, Rp, mask, Cc);

    // ---- stage 3: output projection ----
    out_gemm_kernel<<<dim3(256, 8), blk32, 0, stream>>>(Cc, Wob, bo, out);
}
